// DynamicGAT_44135083934280
// MI455X (gfx1250) — matrix operand order repaired and checked by reading, not compiled, not run on hardware
//
#include <hip/hip_runtime.h>

#define N_NODES 8192
#define N_EDGES 262144
#define EN (N_EDGES + N_NODES)      // edges + self loops
#define IN_C 128
#define HID 64
#define HEADS 4
#define HC (HEADS * HID)            // 256
#define OUTC 16
#define SLOPE 0.2f
#define TH2 0.25f                   // dist < 0.5  <=>  d2 < 0.25

typedef _Float16 v8h  __attribute__((ext_vector_type(8)));
typedef _Float16 v16h __attribute__((ext_vector_type(16)));
typedef float    v8f  __attribute__((ext_vector_type(8)));

__device__ __forceinline__ float lrelu(float x) { return x > 0.f ? x : SLOPE * x; }

// order-preserving float <-> uint for atomicMax-based segment max
__device__ __forceinline__ unsigned enc_f(float f) {
    unsigned u = __float_as_uint(f);
    return (u & 0x80000000u) ? ~u : (u | 0x80000000u);
}
__device__ __forceinline__ float dec_f(unsigned u) {
    u = (u & 0x80000000u) ? (u & 0x7fffffffu) : ~u;
    return __uint_as_float(u);
}

// ---------------- stage 1: h = x @ W1  (8192x128x256, 0.54 GFLOP) ----------------
__global__ void k_gemm1(const float* __restrict__ x, const float* __restrict__ W1,
                        float* __restrict__ h) {
    int idx = blockIdx.x * blockDim.x + threadIdx.x;   // n*256 + m
    int n = idx >> 8, m = idx & 255;
    const float* xr = x + n * IN_C;
    float s = 0.f;
#pragma unroll 8
    for (int k = 0; k < IN_C; ++k) s = fmaf(xr[k], W1[k * HC + m], s);
    h[idx] = s;
}

// ---------------- a_s / a_d per (node, head) ----------------
__global__ void k_attncoef(const float* __restrict__ h, const float* __restrict__ as1,
                           const float* __restrict__ ad1, float* __restrict__ a_s,
                           float* __restrict__ a_d) {
    int idx = blockIdx.x * blockDim.x + threadIdx.x;
    if (idx >= N_NODES * HEADS) return;
    int n = idx >> 2, hd = idx & 3;
    const float* hr = h + n * HC + hd * HID;
    const float* s1 = as1 + hd * HID;
    const float* d1 = ad1 + hd * HID;
    float ss = 0.f, dd = 0.f;
#pragma unroll 8
    for (int c = 0; c < HID; ++c) { ss = fmaf(hr[c], s1[c], ss); dd = fmaf(hr[c], d1[c], dd); }
    a_s[idx] = ss; a_d[idx] = dd;
}

// ---------------- edge pass 1: segment max via encoded atomicMax ----------------
__global__ void k_edge_max(const int* __restrict__ ei, const float* __restrict__ a_s,
                           const float* __restrict__ a_d, unsigned* __restrict__ emax) {
    int e = blockIdx.x * blockDim.x + threadIdx.x;
    if (e >= EN) return;
    int s = (e < N_EDGES) ? ei[e] : (e - N_EDGES);
    int d = (e < N_EDGES) ? ei[N_EDGES + e] : (e - N_EDGES);
#pragma unroll
    for (int hd = 0; hd < HEADS; ++hd) {
        float v = lrelu(a_s[s * HEADS + hd] + a_d[d * HEADS + hd]);
        atomicMax(&emax[d * HEADS + hd], enc_f(v));
    }
}

// ---------------- edge pass 2: ez = exp(e - max), denom = segment sum ----------------
__global__ void k_edge_expsum(const int* __restrict__ ei, const float* __restrict__ a_s,
                              const float* __restrict__ a_d, const unsigned* __restrict__ emax,
                              float* __restrict__ ez, float* __restrict__ denom) {
    int e = blockIdx.x * blockDim.x + threadIdx.x;
    if (e >= EN) return;
    int s = (e < N_EDGES) ? ei[e] : (e - N_EDGES);
    int d = (e < N_EDGES) ? ei[N_EDGES + e] : (e - N_EDGES);
#pragma unroll
    for (int hd = 0; hd < HEADS; ++hd) {
        float v = lrelu(a_s[s * HEADS + hd] + a_d[d * HEADS + hd]);
        float m = dec_f(emax[d * HEADS + hd]);
        float z = __expf(v - m);
        ez[e * HEADS + hd] = z;
        atomicAdd(&denom[d * HEADS + hd], z);
    }
}

// ---------------- edge pass 3: agg[dst] += h[src] * alpha  (block per edge) ----------------
__global__ void k_edge_agg(const int* __restrict__ ei, const float* __restrict__ h,
                           const float* __restrict__ ez, const float* __restrict__ denom,
                           float* __restrict__ agg) {
    int e = blockIdx.x;
    int c = threadIdx.x;                       // 0..255
    int s = (e < N_EDGES) ? ei[e] : (e - N_EDGES);
    int d = (e < N_EDGES) ? ei[N_EDGES + e] : (e - N_EDGES);
    int hd = c >> 6;
    float al = ez[e * HEADS + hd] / denom[d * HEADS + hd];
    atomicAdd(&agg[d * HC + c], h[s * HC + c] * al);
}

// ---------------- h1 = elu(agg + b1); fh = f16(h1); sq = sum(fh^2) ----------------
__global__ void k_h1(const float* __restrict__ agg, const float* __restrict__ b1,
                     float* __restrict__ h1, _Float16* __restrict__ fh,
                     float* __restrict__ sq) {
    int n = blockIdx.x, c = threadIdx.x;       // blockDim = 256
    float v = agg[n * HC + c] + b1[c];
    v = v > 0.f ? v : (__expf(v) - 1.f);
    h1[n * HC + c] = v;
    _Float16 hv = (_Float16)v;
    fh[n * HC + c] = hv;
    float vf = (float)hv;
    __shared__ float red[HC];
    red[c] = vf * vf;
    __syncthreads();
    for (int off = HC / 2; off > 0; off >>= 1) {
        if (c < off) red[c] += red[c + off];
        __syncthreads();
    }
    if (c == 0) sq[n] = red[0];
}

// ---------------- h2 = h1 @ W2 (8192x256x16) ----------------
__global__ void k_h2(const float* __restrict__ h1, const float* __restrict__ W2,
                     float* __restrict__ h2) {
    int idx = blockIdx.x * blockDim.x + threadIdx.x;   // n*16 + o
    int n = idx >> 4, o = idx & 15;
    const float* hr = h1 + n * HC;
    float s = 0.f;
#pragma unroll 8
    for (int k = 0; k < HC; ++k) s = fmaf(hr[k], W2[k * OUTC + o], s);
    h2[idx] = s;
}

__global__ void k_s2t2(const float* __restrict__ h2, const float* __restrict__ as2,
                       const float* __restrict__ ad2, float* __restrict__ s2,
                       float* __restrict__ t2) {
    int n = blockIdx.x * blockDim.x + threadIdx.x;
    if (n >= N_NODES) return;
    float ss = 0.f, tt = 0.f;
#pragma unroll
    for (int o = 0; o < OUTC; ++o) { ss = fmaf(h2[n * OUTC + o], as2[o], ss);
                                     tt = fmaf(h2[n * OUTC + o], ad2[o], tt); }
    s2[n] = ss; t2[n] = tt;
}

// ---------------- fused N^2 kernel: WMMA Gram tiles + masked flash-softmax ----------------
// 512 blocks x 128 threads (4 waves). Each block owns a 16-row i-block; the 4 waves
// split the 512 j-tiles and merge online-softmax states through LDS at the end.
// sched_barrier(0) pins the 16 global_load_b128 ABOVE the WMMA chain so the machine
// scheduler cannot sink them back into a load->wait(0)->wmma lockstep: all 16 loads
// stay in flight and the waitcnt pass emits progressive s_wait_loadcnt thresholds.
__global__ __launch_bounds__(128) void k_fused(
        const _Float16* __restrict__ fh, const float* __restrict__ sq,
        const float* __restrict__ s2, const float* __restrict__ t2,
        const float* __restrict__ h2, const float* __restrict__ b2,
        float* __restrict__ out) {
    const int wave = threadIdx.x >> 5;
    const int lane = threadIdx.x & 31;
    const int rr   = lane & 15;
    const int half = lane >> 4;
    const int i0   = blockIdx.x * 16;

    __shared__ float gtile[4][16][17];
    __shared__ float smx[4][16];
    __shared__ float slm[4][16];
    __shared__ float sacc[4][16][16];
    __shared__ float pre[16][17];

    // resident A fragments: rows i0..i0+15, K=256 -> 8 steps of 32
    // v16h A layout (16x32 f16): lane<16 holds row rr, k={0..15}; lane>=16 k={16..31}
    v16h A[8];
    {
        const _Float16* ar = fh + (size_t)(i0 + rr) * HC;
#pragma unroll
        for (int s = 0; s < 8; ++s) {
            const _Float16* p = ar + s * 32 + half * 16;
            v8h lo = *(const v8h*)p;
            v8h hi = *(const v8h*)(p + 8);
            A[s] = __builtin_shufflevector(lo, hi, 0, 1, 2, 3, 4, 5, 6, 7,
                                                   8, 9, 10, 11, 12, 13, 14, 15);
        }
    }
    float sqRow[8];
#pragma unroll
    for (int r = 0; r < 8; ++r) sqRow[r] = sq[i0 + r + 8 * half];
    const float sqi = sq[i0 + rr];
    const float t2i = t2[i0 + rr];

    float m = -1e30f, lsum = 0.f;
    float acc[8];
#pragma unroll
    for (int c = 0; c < 8; ++c) acc[c] = 0.f;

    for (int jt = wave; jt < N_NODES / 16; jt += 4) {
        const int j0 = jt * 16;
        const _Float16* br = fh + (size_t)(j0 + rr) * HC + half * 16;
        if (jt + 4 < N_NODES / 16)
            __builtin_prefetch(fh + (size_t)((jt + 4) * 16 + rr) * HC + half * 16, 0, 1);

        // batch-load all 8 B fragments (16x global_load_b128, one clause, all in flight)
        v16h B[8];
#pragma unroll
        for (int s = 0; s < 8; ++s) B[s] = *(const v16h*)(br + s * 32);

        float sqJ = sq[j0 + rr];              // issued alongside the B clause

#if __has_builtin(__builtin_amdgcn_sched_barrier)
        __builtin_amdgcn_sched_barrier(0);    // loads may not sink below this point
#endif

        v8f C = {};
#pragma unroll
        for (int s = 0; s < 8; ++s)
            C = __builtin_amdgcn_wmma_f32_16x16x32_f16(false, A[s], false, B[s],
                                                       (short)0, C, false, false);

        // fast adjacency screen straight from accumulator registers
        bool anyadj = false;
#pragma unroll
        for (int r = 0; r < 8; ++r) {
            float d2 = sqRow[r] + sqJ - 2.f * C[r];
            anyadj |= (d2 < TH2);
        }
        if (__any((int)anyadj)) {
            // spill tile: lane (half,rr) reg r holds G[r + 8*half][rr]
#pragma unroll
            for (int r = 0; r < 8; ++r) gtile[wave][r + 8 * half][rr] = C[r];
            __threadfence_block();     // wave-local LDS ordering (no cross-wave barrier!)
#pragma unroll 4
            for (int j = 0; j < 16; ++j) {
                float g  = gtile[wave][rr][j];
                float d2 = sqi + sq[j0 + j] - 2.f * g;
                if (d2 < TH2) {
                    float sc = lrelu(s2[j0 + j] + t2i);
                    float mn = fmaxf(m, sc);
                    float ea = __expf(m - mn), eb = __expf(sc - mn);
                    lsum = lsum * ea + eb;
                    const float* h2j = h2 + (size_t)(j0 + j) * OUTC + half * 8;
#pragma unroll
                    for (int c = 0; c < 8; ++c) acc[c] = fmaf(acc[c], ea, eb * h2j[c]);
                    m = mn;
                }
            }
            __threadfence_block();
        }
    }

    // merge per-wave flash states, then bias + log_softmax over 16 channels
    smx[wave][rr] = m;
    slm[wave][rr] = lsum;
#pragma unroll
    for (int c = 0; c < 8; ++c) sacc[wave][rr][half * 8 + c] = acc[c];
    __syncthreads();

    if (wave == 0) {
        float M = -1e30f;
#pragma unroll
        for (int w = 0; w < 4; ++w) M = fmaxf(M, smx[w][rr]);
        float L = 0.f, acm[8];
#pragma unroll
        for (int c = 0; c < 8; ++c) acm[c] = 0.f;
#pragma unroll
        for (int w = 0; w < 4; ++w) {
            float f = __expf(smx[w][rr] - M);
            L += slm[w][rr] * f;
#pragma unroll
            for (int c = 0; c < 8; ++c) acm[c] += sacc[w][rr][half * 8 + c] * f;
        }
        float inv = 1.f / L;
#pragma unroll
        for (int c = 0; c < 8; ++c) pre[rr][half * 8 + c] = acm[c] * inv + b2[half * 8 + c];
        __threadfence_block();
        if (lane < 16) {
            float mx = -1e30f;
#pragma unroll
            for (int c = 0; c < OUTC; ++c) mx = fmaxf(mx, pre[lane][c]);
            float s = 0.f;
#pragma unroll
            for (int c = 0; c < OUTC; ++c) s += __expf(pre[lane][c] - mx);
            float lse = mx + __logf(s);
            float* o = out + (size_t)(i0 + lane) * OUTC;
#pragma unroll
            for (int c = 0; c < OUTC; ++c) o[c] = pre[lane][c] - lse;
        }
    }
}

extern "C" void kernel_launch(void* const* d_in, const int* in_sizes, int n_in,
                              void* d_out, int out_size, void* d_ws, size_t ws_size,
                              hipStream_t stream) {
    const float* x   = (const float*)d_in[0];
    const int*   ei  = (const int*)  d_in[1];
    const float* W1  = (const float*)d_in[2];
    const float* as1 = (const float*)d_in[3];
    const float* ad1 = (const float*)d_in[4];
    const float* b1  = (const float*)d_in[5];
    const float* W2  = (const float*)d_in[6];
    const float* as2 = (const float*)d_in[7];
    const float* ad2 = (const float*)d_in[8];
    const float* b2  = (const float*)d_in[9];
    float* out = (float*)d_out;

    char* ws = (char*)d_ws;
    size_t off = 0;
    auto alloc = [&](size_t bytes) -> char* {
        char* p = ws + off;
        off = (off + bytes + 255) & ~(size_t)255;
        return p;
    };
    float*    h     = (float*)   alloc((size_t)N_NODES * HC * 4);
    float*    a_s   = (float*)   alloc((size_t)N_NODES * HEADS * 4);
    float*    a_d   = (float*)   alloc((size_t)N_NODES * HEADS * 4);
    unsigned* emax  = (unsigned*)alloc((size_t)N_NODES * HEADS * 4);
    float*    denom = (float*)   alloc((size_t)N_NODES * HEADS * 4);
    float*    ez    = (float*)   alloc((size_t)EN * HEADS * 4);
    float*    agg   = (float*)   alloc((size_t)N_NODES * HC * 4);
    float*    h1    = (float*)   alloc((size_t)N_NODES * HC * 4);
    _Float16* fh    = (_Float16*)alloc((size_t)N_NODES * HC * 2);
    float*    sqv   = (float*)   alloc((size_t)N_NODES * 4);
    float*    h2    = (float*)   alloc((size_t)N_NODES * OUTC * 4);
    float*    s2    = (float*)   alloc((size_t)N_NODES * 4);
    float*    t2    = (float*)   alloc((size_t)N_NODES * 4);

    (void)hipMemsetAsync(emax,  0, (size_t)N_NODES * HEADS * 4, stream);
    (void)hipMemsetAsync(denom, 0, (size_t)N_NODES * HEADS * 4, stream);
    (void)hipMemsetAsync(agg,   0, (size_t)N_NODES * HC * 4, stream);

    k_gemm1   <<<N_NODES * HC / 256, 256, 0, stream>>>(x, W1, h);
    k_attncoef<<<(N_NODES * HEADS + 255) / 256, 256, 0, stream>>>(h, as1, ad1, a_s, a_d);
    k_edge_max<<<(EN + 255) / 256, 256, 0, stream>>>(ei, a_s, a_d, emax);
    k_edge_expsum<<<(EN + 255) / 256, 256, 0, stream>>>(ei, a_s, a_d, emax, ez, denom);
    k_edge_agg<<<EN, 256, 0, stream>>>(ei, h, ez, denom, agg);
    k_h1      <<<N_NODES, HC, 0, stream>>>(agg, b1, h1, fh, sqv);
    k_h2      <<<N_NODES * OUTC / 256, 256, 0, stream>>>(h1, W2, h2);
    k_s2t2    <<<(N_NODES + 255) / 256, 256, 0, stream>>>(h2, as2, ad2, s2, t2);
    k_fused   <<<N_NODES / 16, 128, 0, stream>>>(fh, sqv, s2, t2, h2, b2, out);
}
